// ContrastivePreTraining_65000035057685
// MI455X (gfx1250) — matrix operand order repaired and checked by reading, not compiled, not run on hardware
//
#include <hip/hip_runtime.h>
#include <hip/hip_bf16.h>

// ---------------------------------------------------------------------------
// CGCNN forward on MI455X (gfx1250, wave32, WMMA + async global->LDS).
//
// N=65536 nodes, E=1048576 edges, G=1024 graphs, D=64, Z=192, H=128, NCONV=3.
//
// Edge kernel v2:
//  * persistent blocks (grid-stride over 32-edge tile-pairs)
//  * each wave owns ONE 16-column N-tile; its 12 weight fragments
//    (6 K-steps x {gate, signal}) live in 96 VGPRs for the whole kernel
//  * x[dst]/x[src] gathers staged to LDS as f16; edge_attr (the contiguous
//    HBM stream) is DMA'd to LDS with global_load_async_to_lds_b128 and
//    fenced with s_wait_asynccnt (ASYNCcnt path, inline asm)
//  * 12 v_wmma_f32_16x16x32_f16 per wave-iteration, f32 accumulate,
//    sigmoid*softplus epilogue + global_atomic_add_f32 scatter
// ---------------------------------------------------------------------------

typedef __attribute__((ext_vector_type(16))) _Float16 v16h;
typedef __attribute__((ext_vector_type(8)))  _Float16 v8h;
typedef __attribute__((ext_vector_type(4)))  _Float16 v4h;
typedef __attribute__((ext_vector_type(8)))  float    v8f;

#define DD    64
#define ZZ    192
#define HH    128
#define NCONV 3
#define EDGE_BLOCKS 2048

__device__ __forceinline__ float softplusf(float v) {
    return (v > 20.f) ? v : __logf(1.f + __expf(v));
}

// generic shared pointer -> wave-relative LDS byte offset (low 32 bits)
__device__ __forceinline__ unsigned lds_off(const void* p) {
    return (unsigned)(unsigned long long)p;
}

// ---------------------------------------------------------------- utilities
__global__ void k_zero(float* __restrict__ p, long n) {
    long i = (long)blockIdx.x * blockDim.x + threadIdx.x;
    if (i < n) p[i] = 0.f;
}

// Convert Wf/Ws [NCONV, Z, D] f32 -> transposed f16 [NCONV, D(n), Z(k)]
__global__ void k_prep_w(const float* __restrict__ Wf, const float* __restrict__ Ws,
                         _Float16* __restrict__ wfT, _Float16* __restrict__ wsT) {
    int i = blockIdx.x * blockDim.x + threadIdx.x;
    if (i >= NCONV * DD * ZZ) return;
    int l = i / (DD * ZZ);
    int r = i % (DD * ZZ);
    int n = r / ZZ;
    int k = r % ZZ;
    wfT[i] = (_Float16)Wf[(long)l * ZZ * DD + (long)k * DD + n];
    wsT[i] = (_Float16)Ws[(long)l * ZZ * DD + (long)k * DD + n];
}

__global__ void k_embed(const int* __restrict__ atom_idx, const float* __restrict__ emb,
                        float* __restrict__ x, long n) {
    long i = (long)blockIdx.x * blockDim.x + threadIdx.x;   // over N*D
    if (i >= n) return;
    long node = i >> 6;
    int  d    = (int)(i & 63);
    x[i] = emb[(long)atom_idx[node] * DD + d];
}

// ------------------------------------------------------- WMMA edge messages
__global__ __launch_bounds__(256)
void k_edge_wmma(const float* __restrict__ x,
                 const float* __restrict__ ea,
                 const int* __restrict__ src,
                 const int* __restrict__ dst,
                 const _Float16* __restrict__ wfT,   // layer slice [64][192] f16
                 const _Float16* __restrict__ wsT,
                 const float* __restrict__ bfl,      // [64]
                 const float* __restrict__ bsl,
                 float* __restrict__ agg,
                 long nPairs)                        // E / 32
{
    __shared__ _Float16 sA[32][128];                 // [x_dst | x_src] f16, 8 KB
    __shared__ float    sEA[32][DD];                 // edge_attr f32, 8 KB (async)

    const int wave = threadIdx.x >> 5;
    const int lane = threadIdx.x & 31;

    // CDNA5 16-bit A 16x32 layout: lanes0-15 K in {0..7,16..23}; lanes16-31 +8
    const int m    = lane & 15;
    const int kb   = (lane < 16) ? 0 : 8;
    // CDNA5 16-bit B 32x16 layout: lanes0-15 K in {0..7,16..23}; lanes16-31 +8
    const int koff = (lane < 16) ? 0 : 8;
    const int nl   = lane & 15;
    const int nt   = wave & 3;                       // this wave's 16-col N-tile
    const int tile = wave >> 2;                      // edge sub-tile 0/1
    const int rb   = tile * 16;                      // row base in LDS tiles
    const int mOff = (lane < 16) ? 0 : 8;            // C/D row group
    const int c    = nt * 16 + nl;                   // output column
    const float bfv = bfl[c];
    const float bsv = bsl[c];

    // -------- hoist ALL weight fragments into registers (invariant over E)
    v16h Bf[6], Bs[6];
    #pragma unroll
    for (int s = 0; s < 6; ++s) {
        const _Float16* bp = wfT + (long)(nt * 16 + nl) * ZZ + s * 32 + koff;
        v8h lo = *(const v8h*)bp;
        v8h hi = *(const v8h*)(bp + 16);
        Bf[s] = __builtin_shufflevector(lo, hi,
                0,1,2,3,4,5,6,7,8,9,10,11,12,13,14,15);
        const _Float16* sp = wsT + (long)(nt * 16 + nl) * ZZ + s * 32 + koff;
        v8h slo = *(const v8h*)sp;
        v8h shi = *(const v8h*)(sp + 16);
        Bs[s] = __builtin_shufflevector(slo, shi,
                0,1,2,3,4,5,6,7,8,9,10,11,12,13,14,15);
    }

    for (long tp = blockIdx.x; tp < nPairs; tp += gridDim.x) {
        const long e0 = tp * 32;                     // block's 32 edges

        // ---- async DMA the contiguous edge_attr tile (8 KB) into LDS (f32)
        {
            const unsigned long long gbase =
                (unsigned long long)(const void*)(ea + e0 * DD);
            const unsigned lbase = lds_off(&sEA[0][0]);
            #pragma unroll
            for (int cch = 0; cch < 2; ++cch) {      // 512 x 16B chunks / 256 thr
                int idx = threadIdx.x * 2 + cch;
                unsigned           loff  = lbase + (unsigned)idx * 16u;
                unsigned long long gaddr = gbase + (unsigned long long)idx * 16u;
                asm volatile("global_load_async_to_lds_b128 %0, %1, off"
                             :: "v"(loff), "v"(gaddr) : "memory");
            }
        }

        // ---- gather-stage x[dst] | x[src] rows as f16 (overlaps the DMA)
        #pragma unroll
        for (int it = 0; it < 4; ++it) {
            int idx = it * 256 + threadIdx.x;        // 32 rows x 32 float4
            int row = idx >> 5;
            int c4  = idx & 31;
            long e  = e0 + row;
            const float* p = (c4 < 16) ? (x + (long)dst[e] * DD)
                                       : (x + (long)src[e] * DD);
            float4 v = ((const float4*)p)[c4 & 15];
            v4h h = { (_Float16)v.x, (_Float16)v.y, (_Float16)v.z, (_Float16)v.w };
            *(v4h*)&sA[row][c4 * 4] = h;
        }

        asm volatile("s_wait_asynccnt 0" ::: "memory");  // this wave's DMA done
        __syncthreads();                                 // all waves' staging done

        // ---- 12 WMMAs: zf/zs accumulate over Z=192 (4 f16 steps + 2 ea steps)
        v8f accF, accS;
        #pragma unroll
        for (int r = 0; r < 8; ++r) { accF[r] = 0.f; accS[r] = 0.f; }

        #pragma unroll
        for (int s = 0; s < 4; ++s) {                // K over x_dst|x_src
            const _Float16* ar = &sA[rb + m][s * 32 + kb];
            v8h alo = *(const v8h*)ar;               // ds_load_b128
            v8h ahi = *(const v8h*)(ar + 16);
            v16h A = __builtin_shufflevector(alo, ahi,
                     0,1,2,3,4,5,6,7,8,9,10,11,12,13,14,15);
            accF = __builtin_amdgcn_wmma_f32_16x16x32_f16(
                false, A, false, Bf[s], (short)0, accF, false, false);
            accS = __builtin_amdgcn_wmma_f32_16x16x32_f16(
                false, A, false, Bs[s], (short)0, accS, false, false);
        }
        #pragma unroll
        for (int s = 4; s < 6; ++s) {                // K over edge_attr (f32->f16)
            const float* er = &sEA[rb + m][(s - 4) * 32 + kb];
            float4 a0 = *(const float4*)er;          // ds_load_b128 x4
            float4 a1 = *(const float4*)(er + 4);
            float4 a2 = *(const float4*)(er + 16);
            float4 a3 = *(const float4*)(er + 20);
            v16h A = { (_Float16)a0.x, (_Float16)a0.y, (_Float16)a0.z, (_Float16)a0.w,
                       (_Float16)a1.x, (_Float16)a1.y, (_Float16)a1.z, (_Float16)a1.w,
                       (_Float16)a2.x, (_Float16)a2.y, (_Float16)a2.z, (_Float16)a2.w,
                       (_Float16)a3.x, (_Float16)a3.y, (_Float16)a3.z, (_Float16)a3.w };
            accF = __builtin_amdgcn_wmma_f32_16x16x32_f16(
                false, A, false, Bf[s], (short)0, accF, false, false);
            accS = __builtin_amdgcn_wmma_f32_16x16x32_f16(
                false, A, false, Bs[s], (short)0, accS, false, false);
        }

        // ---- epilogue: bias, sigmoid*softplus, scatter-add to agg
        // C/D layout: VGPR r holds (M = r + mOff, N = c)
        const long eT = e0 + rb;
        int nodes[8];
        #pragma unroll
        for (int r = 0; r < 8; ++r) nodes[r] = dst[eT + mOff + r];

        #pragma unroll
        for (int r = 0; r < 8; ++r) {
            float f  = accF[r] + bfv;
            float g  = accS[r] + bsv;
            float sg = 1.f / (1.f + __expf(-f));
            atomicAdd(&agg[(long)nodes[r] * DD + c], sg * softplusf(g));
        }
        __syncthreads();                             // LDS reuse next iteration
    }
}

// scalar fallback for E % 32 (not exercised with E = 2^20, kept for safety)
__global__ void k_edge_tail(const float* __restrict__ x, const float* __restrict__ ea,
                            const int* __restrict__ src, const int* __restrict__ dst,
                            const float* __restrict__ Wfl, const float* __restrict__ Wsl,
                            const float* __restrict__ bfl, const float* __restrict__ bsl,
                            float* __restrict__ agg, long e0, long E)
{
    long i = (long)blockIdx.x * blockDim.x + threadIdx.x;
    long e = e0 + (i >> 6);
    int  c = (int)(i & 63);
    if (e >= E) return;
    const float* xi = x + (long)dst[e] * DD;
    const float* xj = x + (long)src[e] * DD;
    const float* ep = ea + e * DD;
    float zf = bfl[c], zs = bsl[c];
    for (int k = 0; k < 64; ++k) { zf += xi[k] * Wfl[k * DD + c];          zs += xi[k] * Wsl[k * DD + c]; }
    for (int k = 0; k < 64; ++k) { zf += xj[k] * Wfl[(64 + k) * DD + c];   zs += xj[k] * Wsl[(64 + k) * DD + c]; }
    for (int k = 0; k < 64; ++k) { zf += ep[k] * Wfl[(128 + k) * DD + c];  zs += ep[k] * Wsl[(128 + k) * DD + c]; }
    float sg = 1.f / (1.f + __expf(-zf));
    atomicAdd(&agg[(long)dst[e] * DD + c], sg * softplusf(zs));
}

// ----------------------------------------------------------- batchnorm bits
__global__ void k_colstats(const float* __restrict__ agg, int n,
                           float* __restrict__ sum, float* __restrict__ sumsq) {
    __shared__ float s1[256], s2[256];
    int d = blockIdx.x;              // one block per feature column
    int t = threadIdx.x;
    float a = 0.f, b = 0.f;
    for (int r = t; r < n; r += 256) {
        float v = agg[(long)r * DD + d];
        a += v; b += v * v;
    }
    s1[t] = a; s2[t] = b;
    __syncthreads();
    for (int s = 128; s > 0; s >>= 1) {
        if (t < s) { s1[t] += s1[t + s]; s2[t] += s2[t + s]; }
        __syncthreads();
    }
    if (t == 0) { sum[d] = s1[0]; sumsq[d] = s2[0]; }
}

__global__ void k_finstats(const float* __restrict__ sum, const float* __restrict__ sumsq,
                           float* __restrict__ mu, float* __restrict__ rstd, float n) {
    int d = threadIdx.x;
    if (d >= DD) return;
    float m = sum[d] / n;
    float v = sumsq[d] / n - m * m;
    mu[d]   = m;
    rstd[d] = rsqrtf(v + 1e-5f);
}

__global__ void k_bnres(float* __restrict__ x, const float* __restrict__ agg,
                        const float* __restrict__ mu, const float* __restrict__ rstd,
                        const float* __restrict__ gam, const float* __restrict__ bet, long n) {
    long i = (long)blockIdx.x * blockDim.x + threadIdx.x;
    if (i >= n) return;
    int d = (int)(i & 63);
    x[i] += (agg[i] - mu[d]) * rstd[d] * gam[d] + bet[d];
}

// ------------------------------------------------------------- pooling, MLP
__global__ void k_pool(const float* __restrict__ x, const int* __restrict__ batch,
                       float* __restrict__ crys, float* __restrict__ cnt, long n) {
    long i = (long)blockIdx.x * blockDim.x + threadIdx.x;   // over N*D
    if (i >= n) return;
    long node = i >> 6;
    int  d    = (int)(i & 63);
    int  b    = batch[node];
    atomicAdd(&crys[(long)b * DD + d], x[i]);
    if (d == 0) atomicAdd(&cnt[b], 1.f);
}

__global__ void k_pooldiv_sp(float* __restrict__ crys, const float* __restrict__ cnt,
                             float* __restrict__ spc, long n) {
    long i = (long)blockIdx.x * blockDim.x + threadIdx.x;   // over G*D
    if (i >= n) return;
    float c = cnt[i >> 6];
    spc[i] = softplusf(crys[i] / fmaxf(c, 1.f));
}

__global__ void k_mlp1(const float* __restrict__ spc, const float* __restrict__ W1,
                       const float* __restrict__ b1, float* __restrict__ sph, long n) {
    long i = (long)blockIdx.x * blockDim.x + threadIdx.x;   // over G*H
    if (i >= n) return;
    long g = i >> 7;
    int  j = (int)(i & 127);
    float s = b1[j];
    const float* row = spc + g * DD;
    #pragma unroll 8
    for (int d = 0; d < DD; ++d) s += row[d] * W1[d * HH + j];
    sph[i] = softplusf(s);                                  // softplus for next layer
}

__global__ void k_mlp2(const float* __restrict__ sph, const float* __restrict__ W2,
                       const float* __restrict__ b2, float* __restrict__ out, long n) {
    long i = (long)blockIdx.x * blockDim.x + threadIdx.x;   // over G*H
    if (i >= n) return;
    long g = i >> 7;
    int  j = (int)(i & 127);
    float s = b2[j];
    const float* row = sph + g * HH;
    #pragma unroll 8
    for (int k = 0; k < HH; ++k) s += row[k] * W2[k * HH + j];
    out[i] = s;
}

// ---------------------------------------------------------------- launcher
static inline char* carve(char*& p, size_t bytes) {
    char* r = p;
    p += (bytes + 255) & ~(size_t)255;
    return r;
}

extern "C" void kernel_launch(void* const* d_in, const int* in_sizes, int n_in,
                              void* d_out, int out_size, void* d_ws, size_t ws_size,
                              hipStream_t stream) {
    const int*   atom_idx = (const int*)d_in[0];
    const int*   ei       = (const int*)d_in[1];     // [2, E]: row0=src, row1=dst
    const int*   batch    = (const int*)d_in[2];
    const float* edge_attr= (const float*)d_in[3];
    const float* emb      = (const float*)d_in[4];
    const float* Wf       = (const float*)d_in[5];
    const float* bf       = (const float*)d_in[6];
    const float* Ws       = (const float*)d_in[7];
    const float* bs       = (const float*)d_in[8];
    const float* gamma    = (const float*)d_in[9];
    const float* beta     = (const float*)d_in[10];
    const float* W1       = (const float*)d_in[11];
    const float* b1       = (const float*)d_in[12];
    const float* W2       = (const float*)d_in[13];
    const float* b2       = (const float*)d_in[14];
    float* out = (float*)d_out;

    const long N = in_sizes[0];
    const long E = in_sizes[1] / 2;
    const long G = out_size / HH;
    const int* src = ei;
    const int* dst = ei + E;

    // workspace carve (~35 MB)
    char* p = (char*)d_ws;
    float*     x    = (float*)    carve(p, (size_t)N * DD * 4);
    float*     agg  = (float*)    carve(p, (size_t)N * DD * 4);
    _Float16*  wfT  = (_Float16*) carve(p, (size_t)NCONV * DD * ZZ * 2);
    _Float16*  wsT  = (_Float16*) carve(p, (size_t)NCONV * DD * ZZ * 2);
    float*     csum = (float*)    carve(p, DD * 4);
    float*     csq  = (float*)    carve(p, DD * 4);
    float*     mu   = (float*)    carve(p, DD * 4);
    float*     rstd = (float*)    carve(p, DD * 4);
    float*     crys = (float*)    carve(p, (size_t)G * DD * 4);
    float*     cnt  = (float*)    carve(p, (size_t)G * 4);
    float*     spc  = (float*)    carve(p, (size_t)G * DD * 4);
    float*     sph  = (float*)    carve(p, (size_t)G * HH * 4);
    (void)ws_size; (void)n_in;

    const long ND = N * DD;

    // weight transpose + f16 convert, node embedding
    k_prep_w<<<(NCONV * DD * ZZ + 255) / 256, 256, 0, stream>>>(Wf, Ws, wfT, wsT);
    k_embed<<<(ND + 255) / 256, 256, 0, stream>>>(atom_idx, emb, x, ND);

    const long nPairs = E / 32;           // 32 edges per block-iteration
    const long eRem   = E % 32;
    const int  eGrid  = (int)((nPairs < EDGE_BLOCKS) ? nPairs : EDGE_BLOCKS);

    for (int l = 0; l < NCONV; ++l) {
        k_zero<<<(ND + 255) / 256, 256, 0, stream>>>(agg, ND);
        if (nPairs > 0) {
            k_edge_wmma<<<eGrid, 256, 0, stream>>>(
                x, edge_attr, src, dst,
                wfT + (long)l * DD * ZZ, wsT + (long)l * DD * ZZ,
                bf + l * DD, bs + l * DD, agg, nPairs);
        }
        if (eRem) {
            k_edge_tail<<<(int)((eRem * 64 + 255) / 256), 256, 0, stream>>>(
                x, edge_attr, src, dst,
                Wf + (long)l * ZZ * DD, Ws + (long)l * ZZ * DD,
                bf + l * DD, bs + l * DD, agg, nPairs * 32, E);
        }
        k_colstats<<<DD, 256, 0, stream>>>(agg, (int)N, csum, csq);
        k_finstats<<<1, 64, 0, stream>>>(csum, csq, mu, rstd, (float)N);
        k_bnres<<<(ND + 255) / 256, 256, 0, stream>>>(
            x, agg, mu, rstd, gamma + l * DD, beta + l * DD, ND);
    }

    // segment-mean pooling + MLP head
    k_zero<<<(int)((G * DD + 255) / 256), 256, 0, stream>>>(crys, G * DD);
    k_zero<<<(int)((G + 255) / 256), 256, 0, stream>>>(cnt, G);
    k_pool<<<(ND + 255) / 256, 256, 0, stream>>>(x, batch, crys, cnt, ND);
    k_pooldiv_sp<<<(int)((G * DD + 255) / 256), 256, 0, stream>>>(crys, cnt, spc, G * DD);
    k_mlp1<<<(int)((G * HH + 255) / 256), 256, 0, stream>>>(spc, W1, b1, sph, G * HH);
    k_mlp2<<<(int)((G * HH + 255) / 256), 256, 0, stream>>>(sph, W2, b2, out, G * HH);
}
